// CrossViewFusion_62508954026607
// MI455X (gfx1250) — matrix operand order repaired and checked by reading, not compiled, not run on hardware
//
#include <hip/hip_runtime.h>
#include <hip/hip_bf16.h>

typedef __attribute__((ext_vector_type(16))) __bf16 v16bf;
typedef __attribute__((ext_vector_type(8)))  float  v8f;
typedef __attribute__((ext_vector_type(4)))  int    v4i;

// Problem constants
constexpr int  Bn  = 8;
constexpr int  C1c = 768;
constexpr int  C2c = 768;
constexpr int  Hs  = 32, Ws = 32;
constexpr int  Nsp = Hs * Ws;      // 1024 spatial positions
constexpr int  H1  = 128, W1 = 128;

union FragU { uint4 u[2]; v16bf v; };
__device__ __forceinline__ v16bf make_frag(uint4 lo, uint4 hi) {
    FragU f; f.u[0] = lo; f.u[1] = hi; return f.v;
}

// ---------------------------------------------------------------------------
// CDNA5 async global->LDS 16-byte copy (ASYNCcnt-tracked) + waits
// Builtin signature (per compiler diagnostic): (v4i AS1*, v4i AS3*, imm, imm)
// ---------------------------------------------------------------------------
__device__ __forceinline__ void cp16_async(const __bf16* g, __bf16* l) {
#if __has_builtin(__builtin_amdgcn_global_load_async_to_lds_b128)
    __builtin_amdgcn_global_load_async_to_lds_b128(
        (__attribute__((address_space(1))) v4i*)g,
        (__attribute__((address_space(3))) v4i*)l, 0, 0);
#else
    unsigned lofs = (unsigned)(unsigned long long)(__attribute__((address_space(3))) __bf16*)l;
    asm volatile("global_load_async_to_lds_b128 %0, %1, off"
                 :: "v"(lofs), "v"(g) : "memory");
#endif
}

template <int CNT>
__device__ __forceinline__ void wait_asynccnt() {
#if __has_builtin(__builtin_amdgcn_s_wait_asynccnt)
    __builtin_amdgcn_s_wait_asynccnt(CNT);
#else
    asm volatile("s_wait_asynccnt %0" :: "i"(CNT) : "memory");
#endif
}

// ---------------------------------------------------------------------------
// 1) Fused mean-pool (4x4) + transpose: x1 [B,C1,128,128] f32 -> x1fT [B,N,C1] bf16
//    Reads coalesced along w (dominant 402MB stream); writes coalesced along c.
//    grid (Nsp/64, C1/64, B), block 256
// ---------------------------------------------------------------------------
__global__ __launch_bounds__(256)
void pool_transpose_kernel(const float* __restrict__ x1, __bf16* __restrict__ x1fT) {
    __shared__ __bf16 tile[64][72];
    const int n0 = blockIdx.x * 64;
    const int c0 = blockIdx.y * 64;
    const long bofs = (long)blockIdx.z * C1c * (H1 * W1);
    const int tn = threadIdx.x & 63;
    const int t4 = threadIdx.x >> 6;      // 0..3
    const int n  = n0 + tn;
    const int w4 = (n & (Ws - 1)) * 4;
    const int h4 = (n >> 5) * 4;
    for (int cc = t4; cc < 64; cc += 4) {
        const float* p = x1 + bofs + (long)(c0 + cc) * (H1 * W1) + h4 * W1 + w4;
        __builtin_prefetch(p + 4 * (H1 * W1), 0, 1);   // next cc iteration
        float s = 0.f;
#pragma unroll
        for (int dh = 0; dh < 4; ++dh) {
            float4 v4 = *reinterpret_cast<const float4*>(p + dh * W1);
            s += v4.x + v4.y + v4.z + v4.w;
        }
        tile[cc][tn] = (__bf16)(s * 0.0625f);
    }
    __syncthreads();
    __bf16* outp = x1fT + (long)blockIdx.z * Nsp * C1c;
    for (int nn = t4; nn < 64; nn += 4)
        outp[(long)(n0 + nn) * C1c + c0 + tn] = tile[tn][nn];
}

// ---------------------------------------------------------------------------
// 2) Fused convert + transpose: x2 [B,C2,N] f32 -> x2fT [B,N,C2] bf16
//    grid (Nsp/64, C2/64, B), block 256
// ---------------------------------------------------------------------------
__global__ __launch_bounds__(256)
void convert_transpose_kernel(const float* __restrict__ in, __bf16* __restrict__ outT) {
    __shared__ __bf16 tile[64][72];
    const int n0 = blockIdx.x * 64;
    const int c0 = blockIdx.y * 64;
    const float* ip = in + (long)blockIdx.z * C2c * Nsp;
    const int tn = threadIdx.x & 63;
    const int t4 = threadIdx.x >> 6;
    for (int cc = t4; cc < 64; cc += 4)
        tile[cc][tn] = (__bf16)ip[(long)(c0 + cc) * Nsp + n0 + tn];
    __syncthreads();
    __bf16* outp = outT + (long)blockIdx.z * Nsp * C2c;
    for (int nn = t4; nn < 64; nn += 4)
        outp[(long)(n0 + nn) * C2c + c0 + tn] = tile[tn][nn];
}

// ---------------------------------------------------------------------------
// 3) fp32 -> bf16 convert for weights (vectorized x4)
// ---------------------------------------------------------------------------
__global__ __launch_bounds__(256)
void f32_to_bf16_kernel(const float* __restrict__ in, __bf16* __restrict__ out, long n4) {
    long i = (long)blockIdx.x * blockDim.x + threadIdx.x;
    if (i >= n4) return;
    float4 v = reinterpret_cast<const float4*>(in)[i];
    union { __bf16 h[4]; uint2 u; } pk;
    pk.h[0] = (__bf16)v.x; pk.h[1] = (__bf16)v.y;
    pk.h[2] = (__bf16)v.z; pk.h[3] = (__bf16)v.w;
    reinterpret_cast<uint2*>(out)[i] = pk.u;
}

// ---------------------------------------------------------------------------
// 4) NT WMMA GEMM, async double-buffered:
//      C[m,n] = sum_k A[m,k] * Bt[n,k]    (A:[M,K], Bt:[Nc,K], row-major bf16)
//    Block tile 128(M) x 64(N), 8 wave32 waves, 2x2 of 16x16x32 bf16 WMMA each.
//    K staged in 32-wide double-buffered LDS tiles via global_load_async_to_lds.
//    MODE 0: bf16 store (OUTT: store transposed [n,m])
//    MODE 1: f32 store * scale
//    MODE 2: f32 store + residual Res
// ---------------------------------------------------------------------------
template <int MODE, bool OUTT>
__global__ __launch_bounds__(256)
void gemm_nt_wmma_kernel(const __bf16* __restrict__ A, const __bf16* __restrict__ Bt,
                         void* __restrict__ Out, const float* __restrict__ Res,
                         int M, int Nc, int K,
                         long strideA, long strideB, long strideO, float scale) {
    constexpr int LDA = 40, LDB = 40;      // 32 + 8 pad halfs (16B-aligned chunks)
    constexpr int ASZ = 128 * LDA, BSZ = 64 * LDB;
    __shared__ __align__(16) __bf16 As[2 * ASZ];
    __shared__ __align__(16) __bf16 Bs[2 * BSZ];

    const __bf16* Ab = A  + (long)blockIdx.z * strideA;
    const __bf16* Bb = Bt + (long)blockIdx.z * strideB;
    const long obase = (long)blockIdx.z * strideO;
    const int m0 = blockIdx.y * 128;
    const int n0 = blockIdx.x * 64;

    const int t = threadIdx.x;
    // copy mapping: A tile 128x32 -> 2 x 16B per thread; B tile 64x32 -> 1 x 16B
    const int arow = t >> 1, ajo = (t & 1) * 16;
    const int brow = t >> 2, bko = (t & 3) * 8;
    const __bf16* agp = Ab + (long)(m0 + arow) * K + ajo;
    const __bf16* bgp = Bb + (long)(n0 + brow) * K + bko;

    const int lane = t & 31, wv = t >> 5;
    const int wm = wv & 3, wn = wv >> 2;
    const int l15 = lane & 15, h = lane >> 4;
    const int h8 = h * 8, h16 = h * 16;

    const v8f vzero = {0.f, 0.f, 0.f, 0.f, 0.f, 0.f, 0.f, 0.f};
    v8f acc[2][2];
#pragma unroll
    for (int i = 0; i < 2; ++i)
#pragma unroll
        for (int j = 0; j < 2; ++j) acc[i][j] = vzero;

    auto fill = [&](int buf, int kt) {
        __bf16* al = &As[buf * ASZ + arow * LDA + ajo];
        cp16_async(agp + kt,     al);
        cp16_async(agp + kt + 8, al + 8);
        cp16_async(bgp + kt, &Bs[buf * BSZ + brow * LDB + bko]);
    };

    fill(0, 0);
    const int iters = K >> 5;
    for (int i = 0; i < iters; ++i) {
        const int cur = i & 1;
        const bool hasNext = (i + 1 < iters);
        if (hasNext) {
            fill(cur ^ 1, (i + 1) << 5);
            wait_asynccnt<3>();            // current buffer's 3 copies done
        } else {
            wait_asynccnt<0>();
        }
        __syncthreads();

        const __bf16* Ac = &As[cur * ASZ];
        const __bf16* Bc = &Bs[cur * BSZ];
        v16bf fa[2], fb[2];
#pragma unroll
        for (int sm = 0; sm < 2; ++sm) {
            int row = wm * 32 + sm * 16 + l15;
            uint4 lo = *reinterpret_cast<const uint4*>(&Ac[row * LDA + h8]);
            uint4 hi = *reinterpret_cast<const uint4*>(&Ac[row * LDA + h8 + 16]);
            fa[sm] = make_frag(lo, hi);
        }
#pragma unroll
        for (int sn = 0; sn < 2; ++sn) {
            int col = wn * 32 + sn * 16 + l15;
            uint4 lo = *reinterpret_cast<const uint4*>(&Bc[col * LDB + h8]);
            uint4 hi = *reinterpret_cast<const uint4*>(&Bc[col * LDB + h8 + 16]);
            fb[sn] = make_frag(lo, hi);
        }
#pragma unroll
        for (int sm = 0; sm < 2; ++sm)
#pragma unroll
            for (int sn = 0; sn < 2; ++sn)
                acc[sm][sn] = __builtin_amdgcn_wmma_f32_16x16x32_bf16(
                    false, fa[sm], false, fb[sn], (short)0, acc[sm][sn], false, false);
        __syncthreads();   // all waves done reading `cur` before its refill
    }

    // epilogue: C/D layout -> VGPR r is row (r + 8*h), col = lane&15
#pragma unroll
    for (int sm = 0; sm < 2; ++sm) {
#pragma unroll
        for (int sn = 0; sn < 2; ++sn) {
            int rowbase = m0 + wm * 32 + sm * 16 + 8 * h;
            int col     = n0 + wn * 32 + sn * 16 + l15;
#pragma unroll
            for (int r = 0; r < 8; ++r) {
                float val = acc[sm][sn][r];
                long idx = OUTT ? (obase + (long)col * M + rowbase + r)
                                : (obase + (long)(rowbase + r) * Nc + col);
                if (MODE == 0) {
                    reinterpret_cast<__bf16*>(Out)[idx] = (__bf16)val;
                } else if (MODE == 1) {
                    reinterpret_cast<float*>(Out)[idx] = val * scale;
                } else {
                    reinterpret_cast<float*>(Out)[idx] = val + Res[idx];
                }
            }
        }
    }
}

// ---------------------------------------------------------------------------
// 5) Row softmax over 768 entries, fp32 in -> bf16 out. One block per row.
// ---------------------------------------------------------------------------
__global__ __launch_bounds__(256)
void softmax_kernel(const float* __restrict__ attn, __bf16* __restrict__ out) {
    const long row = blockIdx.x;
    const float* p = attn + row * (long)C1c;
    const int t = threadIdx.x;
    __shared__ float red[256];

    float v0 = p[t], v1 = p[t + 256], v2 = p[t + 512];
    float m = fmaxf(fmaxf(v0, v1), v2);
    red[t] = m; __syncthreads();
    for (int s = 128; s > 0; s >>= 1) {
        if (t < s) red[t] = fmaxf(red[t], red[t + s]);
        __syncthreads();
    }
    m = red[0]; __syncthreads();

    float e0 = __expf(v0 - m), e1 = __expf(v1 - m), e2 = __expf(v2 - m);
    red[t] = e0 + e1 + e2; __syncthreads();
    for (int s = 128; s > 0; s >>= 1) {
        if (t < s) red[t] += red[t + s];
        __syncthreads();
    }
    float inv = 1.f / red[0];

    __bf16* q = out + row * (long)C1c;
    q[t]       = (__bf16)(e0 * inv);
    q[t + 256] = (__bf16)(e1 * inv);
    q[t + 512] = (__bf16)(e2 * inv);
}

// ---------------------------------------------------------------------------
// Host launch
// ---------------------------------------------------------------------------
extern "C" void kernel_launch(void* const* d_in, const int* in_sizes, int n_in,
                              void* d_out, int out_size, void* d_ws, size_t ws_size,
                              hipStream_t stream) {
    const float* x1 = (const float*)d_in[0];   // [8,768,128,128]
    const float* x2 = (const float*)d_in[1];   // [8,768,32,32]
    const float* Wq = (const float*)d_in[2];   // [768,768]
    const float* Wk = (const float*)d_in[3];
    const float* Wv = (const float*)d_in[4];
    float* out = (float*)d_out;                // [8,768,32,32]

    char* ws = (char*)d_ws;
    const long WN = (long)C1c * C1c;           // 589,824 weight elems

    __bf16* x2fT = (__bf16*)(ws + 0);                 // [B,N,C2] 12,582,912 B
    __bf16* x1fT = (__bf16*)(ws + 12582912);          // [B,N,C1] 12,582,912 B
    __bf16* Wqb  = (__bf16*)(ws + 25165824);          //  1,179,648 B
    __bf16* Wkb  = (__bf16*)(ws + 26345472);
    __bf16* Wvb  = (__bf16*)(ws + 27525120);
    __bf16* qb   = (__bf16*)(ws + 28704768);          // [B,C2,N] 12,582,912 B
    __bf16* kb   = (__bf16*)(ws + 41287680);          // [B,C1,N]
    __bf16* vbT  = (__bf16*)(ws + 53870592);          // [B,N,C1]  end 66,453,504 B
    // aliases into dead regions:
    float*  attn  = (float*)(ws + 0);          // over x2fT+x1fT (dead after q/k/v)
    __bf16* attnp = (__bf16*)(ws + 28704768);  // over q (dead after attn GEMM)

    // 1) pool+transpose x1 -> x1fT ; convert+transpose x2 -> x2fT
    dim3 gT(Nsp / 64, C1c / 64, Bn);           // (16,12,8)
    pool_transpose_kernel<<<gT, 256, 0, stream>>>(x1, x1fT);
    convert_transpose_kernel<<<gT, 256, 0, stream>>>(x2, x2fT);
    // 2) weights -> bf16
    f32_to_bf16_kernel<<<(unsigned)((WN / 4 + 255) / 256), 256, 0, stream>>>(Wq, Wqb, WN / 4);
    f32_to_bf16_kernel<<<(unsigned)((WN / 4 + 255) / 256), 256, 0, stream>>>(Wk, Wkb, WN / 4);
    f32_to_bf16_kernel<<<(unsigned)((WN / 4 + 255) / 256), 256, 0, stream>>>(Wv, Wvb, WN / 4);

    const long sX = (long)C1c * Nsp;   // 786,432 per-batch [C,N]/[N,C] stride
    const long sA = (long)C2c * C1c;   // 589,824 per-batch [C2,C1] stride

    // 3) q = Wq x2fT^T ; k = Wk x1fT^T ; v stored transposed as vT [N,C1]
    dim3 gQ(Nsp / 64, C2c / 128, Bn);  // (16,6,8)
    gemm_nt_wmma_kernel<0, false><<<gQ, 256, 0, stream>>>(Wqb, x2fT, qb, nullptr,
        C2c, Nsp, C2c, 0, sX, sX, 1.f);
    gemm_nt_wmma_kernel<0, false><<<gQ, 256, 0, stream>>>(Wkb, x1fT, kb, nullptr,
        C1c, Nsp, C1c, 0, sX, sX, 1.f);
    gemm_nt_wmma_kernel<0, true><<<gQ, 256, 0, stream>>>(Wvb, x1fT, vbT, nullptr,
        C1c, Nsp, C1c, 0, sX, sX, 1.f);

    // 4) attn = (q @ k^T) / 32   (f32, scaled)
    dim3 gA(C1c / 64, C2c / 128, Bn);  // (12,6,8)
    gemm_nt_wmma_kernel<1, false><<<gA, 256, 0, stream>>>(qb, kb, attn, nullptr,
        C2c, C1c, Nsp, sX, sX, sA, 1.f / 32.f);

    // 5) softmax rows -> bf16 probabilities
    softmax_kernel<<<(unsigned)(Bn * C2c), 256, 0, stream>>>(attn, attnp);

    // 6) out = x2 + attnp @ vT^T   (f32 + residual)
    gemm_nt_wmma_kernel<2, false><<<gQ, 256, 0, stream>>>(attnp, vbT, out, x2,
        C2c, Nsp, C1c, sA, sX, sX, 1.f);

    (void)in_sizes; (void)n_in; (void)out_size; (void)ws_size;
}
